// CSPLayer_69363721830733
// MI455X (gfx1250) — compile-verified, matrix operand order checked by reading, not run on hardware
//
#include <hip/hip_runtime.h>
#include <hip/hip_fp16.h>
#include <hip/hip_bf16.h>

typedef __attribute__((ext_vector_type(16))) _Float16 v16h;
typedef __attribute__((ext_vector_type(8)))  _Float16 v8h;
typedef __attribute__((ext_vector_type(8)))  float    v8f;

#define HH 128
#define NN 50000
#define EE 1000000
#define KT1E 9    /* 268 padded to 288 = 9*32 */
#define KT2  4    /* 128 = 4*32 */
#define KT1N 8    /* 256 = 8*32 */

// ---------------------------------------------------------------------------
// Pack f32 weight matrix [K x 128] into WMMA B-fragment order:
// flat index = (((ntile*KT + kt)*32 + lane)*16 + i), half precision.
// Per the CDNA5 16-bit operand layout: lane<16 -> kbase 0, lane>=16 -> kbase 8;
// element i<8 -> K = kt*32 + kbase + i ; i>=8 -> K = kt*32 + kbase + 16 + (i-8).
// Column = ntile*16 + (lane & 15). K >= Kreal padded with zeros.
// ---------------------------------------------------------------------------
__global__ void pack_kernel(const float* __restrict__ W, int Kreal, int KT,
                            _Float16* __restrict__ out) {
    int tid = blockIdx.x * blockDim.x + threadIdx.x;
    int total = 8 * KT * 32 * 16;
    if (tid >= total) return;
    int i    = tid & 15;
    int lane = (tid >> 4) & 31;
    int kt   = (tid >> 9) % KT;
    int n    = tid / (KT * 512);
    int kbase = (lane < 16) ? 0 : 8;
    int ko = (i < 8) ? (kbase + i) : (kbase + 16 + (i - 8));
    int k  = kt * 32 + ko;
    int col = n * 16 + (lane & 15);
    float v = (k < Kreal) ? W[(long long)k * HH + col] : 0.0f;
    out[tid] = (_Float16)v;
}

// Count edges per source node (scatter-mean denominator).
__global__ void cnt_kernel(const int* __restrict__ src, float* __restrict__ cnt) {
    long long e = (long long)blockIdx.x * blockDim.x + threadIdx.x;
    if (e < EE) unsafeAtomicAdd(&cnt[src[e]], 1.0f);
}

// ---------------------------------------------------------------------------
// One MLP layer, 128 rows x 128 cols, y = silu(x @ W + b), result f16 to LDS.
// Wave w owns N-tile w (cols 16w..16w+15); loops 8 M-tiles, KT K-tiles.
// ---------------------------------------------------------------------------
template<int KT>
__device__ __forceinline__ void wmma_layer_to_lds(
    const _Float16* __restrict__ sIn, int inStride,
    const v16h*     __restrict__ wpack,
    const float*    __restrict__ bias,
    _Float16*       __restrict__ sOut, int outStride)
{
    const int lane  = threadIdx.x & 31;
    const int w     = threadIdx.x >> 5;     // wave id == N-tile
    const int lm    = lane & 15;
    const int hi16  = lane >> 4;
    const int kbase = hi16 ? 8 : 0;

    v16h bf[KT];
#pragma unroll
    for (int k = 0; k < KT; ++k)
        bf[k] = wpack[(w * KT + k) * 32 + lane];

    const int   col = w * 16 + lm;
    const float bc  = bias[col];

#pragma unroll
    for (int m = 0; m < 8; ++m) {
        v8f c = {};
#pragma unroll
        for (int k = 0; k < KT; ++k) {
            const _Float16* p = sIn + (m * 16 + lm) * inStride + k * 32 + kbase;
            union { v16h v; v8h h[2]; } a;
            a.h[0] = *(const v8h*)(p);
            a.h[1] = *(const v8h*)(p + 16);
            c = __builtin_amdgcn_wmma_f32_16x16x32_f16(false, a.v, false, bf[k],
                                                       (short)0, c, false, false);
        }
#pragma unroll
        for (int r = 0; r < 8; ++r) {
            int row = m * 16 + r + hi16 * 8;
            float v = c[r] + bc;
            float s = v / (1.0f + __expf(-v));
            sOut[row * outStride + col] = (_Float16)s;
        }
    }
}

// ---------------------------------------------------------------------------
// Edge kernel: 1 block == 128 edges, 256 threads (8 waves).
// edges_input = [h_src | h_dst | lattice | frac_diff | l_f] (268 -> pad 288)
// 2-layer SiLU MLP via WMMA, then atomic scatter-add into sums (== d_out).
// ---------------------------------------------------------------------------
__global__ __launch_bounds__(256)
void edge_kernel(const float* __restrict__ nf,   const float* __restrict__ frac,
                 const float* __restrict__ lat,  const int* __restrict__ src,
                 const int* __restrict__ dst,    const int* __restrict__ e2g,
                 const float* __restrict__ lf,
                 const v16h* __restrict__ w1p,   const float* __restrict__ b1,
                 const v16h* __restrict__ w2p,   const float* __restrict__ b2,
                 float* __restrict__ sums)
{
    __shared__ _Float16 sX[128 * 296];   // A panel, 288 used + pad
    __shared__ _Float16 sY[128 * 136];   // layer-1 activations
    __shared__ int sSrc[128];
    __shared__ int sDst[128];

    const int t = threadIdx.x;
    const long long eBase = (long long)blockIdx.x * 128;

    // pass 0: indices + tail features (cols 256..287)
    if (t < 128) {
        long long e = eBase + t;
        _Float16* row = &sX[t * 296];
        if (e < EE) {
            int s = src[e], d = dst[e], g = e2g[e];
            sSrc[t] = s; sDst[t] = d;
#pragma unroll
            for (int j = 0; j < 6; ++j) row[256 + j] = (_Float16)lat[g * 6 + j];
#pragma unroll
            for (int j = 0; j < 3; ++j) {
                float fd = frac[(long long)d * 3 + j] - frac[(long long)s * 3 + j];
                fd -= floorf(fd);                       // python-style mod 1.0
                row[262 + j] = (_Float16)fd;
            }
#pragma unroll
            for (int j = 0; j < 3; ++j) row[265 + j] = (_Float16)lf[e * 3 + j];
            for (int j = 268; j < 288; ++j) row[j] = (_Float16)0.0f;
        } else {
            sSrc[t] = -1; sDst[t] = -1;
            for (int j = 256; j < 288; ++j) row[j] = (_Float16)0.0f;
        }
    }
    __syncthreads();

    // pass 1: coalesced float2 gathers of h_src / h_dst (cols 0..255)
    const float2* nf2 = (const float2*)nf;
    for (int idx = t; idx < 128 * 64; idx += 256) {
        int row = idx >> 6;
        int c2  = idx & 63;
        int s = sSrc[row], d = sDst[row];
        float2 a = make_float2(0.0f, 0.0f), b = make_float2(0.0f, 0.0f);
        if (s >= 0) a = nf2[(long long)s * 64 + c2];
        if (d >= 0) b = nf2[(long long)d * 64 + c2];
        _Float16* rp = &sX[row * 296];
        rp[c2 * 2]           = (_Float16)a.x;
        rp[c2 * 2 + 1]       = (_Float16)a.y;
        rp[128 + c2 * 2]     = (_Float16)b.x;
        rp[128 + c2 * 2 + 1] = (_Float16)b.y;
    }
    __syncthreads();

    // layer 1: [128x288] @ [288x128] -> silu -> sY
    wmma_layer_to_lds<KT1E>(sX, 296, w1p, b1, sY, 136);
    __syncthreads();

    // layer 2 + scatter-add epilogue
    const int lane  = t & 31;
    const int w     = t >> 5;
    const int lm    = lane & 15;
    const int hi16  = lane >> 4;
    const int kbase = hi16 ? 8 : 0;
    v16h bf[KT2];
#pragma unroll
    for (int k = 0; k < KT2; ++k) bf[k] = w2p[(w * KT2 + k) * 32 + lane];
    const int   col = w * 16 + lm;
    const float bc  = b2[col];
#pragma unroll
    for (int m = 0; m < 8; ++m) {
        v8f c = {};
#pragma unroll
        for (int k = 0; k < KT2; ++k) {
            const _Float16* p = &sY[(m * 16 + lm) * 136 + k * 32 + kbase];
            union { v16h v; v8h h[2]; } a;
            a.h[0] = *(const v8h*)(p);
            a.h[1] = *(const v8h*)(p + 16);
            c = __builtin_amdgcn_wmma_f32_16x16x32_f16(false, a.v, false, bf[k],
                                                       (short)0, c, false, false);
        }
#pragma unroll
        for (int r = 0; r < 8; ++r) {
            int row = m * 16 + r + hi16 * 8;
            int s = sSrc[row];
            if (s >= 0) {
                float v  = c[r] + bc;
                float sv = v / (1.0f + __expf(-v));
                unsafeAtomicAdd(&sums[(long long)s * HH + col], sv);
            }
        }
    }
}

// ---------------------------------------------------------------------------
// Node kernel: 1 block == 128 nodes. input = [node_features | sums/max(cnt,1)]
// 2-layer SiLU MLP via WMMA + residual, in-place on d_out (sums == out).
// ---------------------------------------------------------------------------
__global__ __launch_bounds__(256)
void node_kernel(const float* __restrict__ nf, const float* __restrict__ sums,
                 const float* __restrict__ cnt,
                 const v16h* __restrict__ w1p, const float* __restrict__ b1,
                 const v16h* __restrict__ w2p, const float* __restrict__ b2,
                 float* __restrict__ out)
{
    __shared__ _Float16 sX[128 * 264];   // 256 used + pad
    __shared__ _Float16 sY[128 * 136];
    __shared__ float sInv[128];

    const int t = threadIdx.x;
    const int nBase = blockIdx.x * 128;

    if (t < 128) {
        int n = nBase + t;
        sInv[t] = (n < NN) ? (1.0f / fmaxf(cnt[n], 1.0f)) : -1.0f;
    }
    __syncthreads();

    const float2* nf2 = (const float2*)nf;
    const float2* sm2 = (const float2*)sums;
    for (int idx = t; idx < 128 * 64; idx += 256) {
        int row = idx >> 6;
        int c2  = idx & 63;
        int n   = nBase + row;
        float inv = sInv[row];
        float2 a = make_float2(0.0f, 0.0f), b = make_float2(0.0f, 0.0f);
        if (inv >= 0.0f) {
            a = nf2[(long long)n * 64 + c2];
            b = sm2[(long long)n * 64 + c2];
            b.x *= inv; b.y *= inv;
        }
        _Float16* rp = &sX[row * 264];
        rp[c2 * 2]           = (_Float16)a.x;
        rp[c2 * 2 + 1]       = (_Float16)a.y;
        rp[128 + c2 * 2]     = (_Float16)b.x;
        rp[128 + c2 * 2 + 1] = (_Float16)b.y;
    }
    __syncthreads();

    wmma_layer_to_lds<KT1N>(sX, 264, w1p, b1, sY, 136);
    __syncthreads();

    // layer 2 + residual store
    const int lane  = t & 31;
    const int w     = t >> 5;
    const int lm    = lane & 15;
    const int hi16  = lane >> 4;
    const int kbase = hi16 ? 8 : 0;
    v16h bf[KT2];
#pragma unroll
    for (int k = 0; k < KT2; ++k) bf[k] = w2p[(w * KT2 + k) * 32 + lane];
    const int   col = w * 16 + lm;
    const float bc  = b2[col];
#pragma unroll
    for (int m = 0; m < 8; ++m) {
        v8f c = {};
#pragma unroll
        for (int k = 0; k < KT2; ++k) {
            const _Float16* p = &sY[(m * 16 + lm) * 136 + k * 32 + kbase];
            union { v16h v; v8h h[2]; } a;
            a.h[0] = *(const v8h*)(p);
            a.h[1] = *(const v8h*)(p + 16);
            c = __builtin_amdgcn_wmma_f32_16x16x32_f16(false, a.v, false, bf[k],
                                                       (short)0, c, false, false);
        }
#pragma unroll
        for (int r = 0; r < 8; ++r) {
            int row = m * 16 + r + hi16 * 8;
            int n   = nBase + row;
            if (n < NN) {
                float v  = c[r] + bc;
                float sv = v / (1.0f + __expf(-v));
                out[(long long)n * HH + col] = nf[(long long)n * HH + col] + sv;
            }
        }
    }
}

// ---------------------------------------------------------------------------
extern "C" void kernel_launch(void* const* d_in, const int* in_sizes, int n_in,
                              void* d_out, int out_size, void* d_ws, size_t ws_size,
                              hipStream_t stream) {
    const float* nf   = (const float*)d_in[0];
    const float* frac = (const float*)d_in[1];
    const float* lat  = (const float*)d_in[2];
    const int*   eidx = (const int*)  d_in[3];   // [2, E]
    const int*   e2g  = (const int*)  d_in[4];
    const float* lf   = (const float*)d_in[5];
    const float* eW1  = (const float*)d_in[6];
    const float* eb1  = (const float*)d_in[7];
    const float* eW2  = (const float*)d_in[8];
    const float* eb2  = (const float*)d_in[9];
    const float* nW1  = (const float*)d_in[10];
    const float* nb1  = (const float*)d_in[11];
    const float* nW2  = (const float*)d_in[12];
    const float* nb2  = (const float*)d_in[13];
    float* out = (float*)d_out;

    // workspace: counts + fragment-packed f16 weights
    char* ws = (char*)d_ws;
    size_t off = 0;
    float* cnt = (float*)(ws + off); off += (((size_t)NN * 4) + 255) & ~(size_t)255;
    _Float16* eW1p = (_Float16*)(ws + off); off += (size_t)8 * KT1E * 512 * 2;
    _Float16* eW2p = (_Float16*)(ws + off); off += (size_t)8 * KT2  * 512 * 2;
    _Float16* nW1p = (_Float16*)(ws + off); off += (size_t)8 * KT1N * 512 * 2;
    _Float16* nW2p = (_Float16*)(ws + off); off += (size_t)8 * KT2  * 512 * 2;

    hipMemsetAsync(out, 0, (size_t)NN * HH * sizeof(float), stream);  // sums accumulator
    hipMemsetAsync(cnt, 0, (size_t)NN * sizeof(float), stream);

    pack_kernel<<<(8 * KT1E * 512 + 255) / 256, 256, 0, stream>>>(eW1, 268, KT1E, eW1p);
    pack_kernel<<<(8 * KT2  * 512 + 255) / 256, 256, 0, stream>>>(eW2, 128, KT2,  eW2p);
    pack_kernel<<<(8 * KT1N * 512 + 255) / 256, 256, 0, stream>>>(nW1, 256, KT1N, nW1p);
    pack_kernel<<<(8 * KT2  * 512 + 255) / 256, 256, 0, stream>>>(nW2, 128, KT2,  nW2p);

    cnt_kernel<<<(EE + 255) / 256, 256, 0, stream>>>(eidx, cnt);

    edge_kernel<<<(EE + 127) / 128, 256, 0, stream>>>(
        nf, frac, lat, eidx, eidx + EE, e2g, lf,
        (const v16h*)eW1p, eb1, (const v16h*)eW2p, eb2, out);

    node_kernel<<<(NN + 127) / 128, 256, 0, stream>>>(
        nf, out, cnt, (const v16h*)nW1p, nb1, (const v16h*)nW2p, nb2, out);
}
